// HCL_37907381354588
// MI455X (gfx1250) — matrix operand order repaired and checked by reading, not compiled, not run on hardware
//
#include <hip/hip_runtime.h>

// ---------------------------------------------------------------------------
// Types for CDNA5 WMMA (wave32): v_wmma_f32_16x16x32_bf16
// ---------------------------------------------------------------------------
typedef __attribute__((ext_vector_type(16))) __bf16 v16bf;
typedef __attribute__((ext_vector_type(8)))  float  v8f;

#define NTOK 4096
#define DIM  512

// round-to-nearest-even f32 -> bf16 bits
__device__ __forceinline__ unsigned short f2bf(float f) {
    unsigned int u = __float_as_uint(f);
    unsigned int r = (u + 0x7FFFu + ((u >> 16) & 1u)) >> 16;
    return (unsigned short)r;
}

union FragU { uint4 u[2]; v16bf v; };

__device__ __forceinline__ v8f wmma_bf16(v16bf a, v16bf b, v8f c) {
    return __builtin_amdgcn_wmma_f32_16x16x32_bf16(false, a, false, b, (short)0, c,
                                                   false, false);
}

// ---------------------------------------------------------------------------
// GEMM: C[M,N] = act(A[M,K] @ W[N,K]^T + bias [+ resid]) ; bf16 in, f32 acc
// Block: 256 thr = 8 waves; wave tile 32x32 (2x2 wmma); block tile 128x64.
// LDS double-buffered tiles (BK=32), padded rows (40 bf16 = 80 B) so all
// ds_load_b128 / ds_store_b128 stay 16B-aligned.
//
// WMMA bf16 fragment lane mappings (ISA 7.12.2):
//   A 16x32: lane L (hi=L>>4, m=L&15): elems 0..7 -> K=hi*8+e ; 8..15 -> K=16+hi*8+(e-8)
//   B 32x16: lane L: column n=L&15, K as A: 0..7 -> hi*8+e ; 8..15 -> 16+hi*8+(e-8)
//   C/D 16x16: lane L, elem r: m = 8*(L>>4)+r, n = L&15
// ---------------------------------------------------------------------------
__global__ __launch_bounds__(256) void gemm_bf16_kernel(
    const unsigned short* __restrict__ A,   // [M,K]
    const unsigned short* __restrict__ W,   // [N,K]
    const float* __restrict__ bias,         // [N] or null
    const float* __restrict__ resid,        // [M,N] or null
    float* __restrict__ out32,              // [M,N] or null
    unsigned short* __restrict__ out16,     // [M,N] (or [N,M] if t16) or null
    int M, int N, int K, int relu, int t16)
{
    __shared__ unsigned short lA[2][128][40];
    __shared__ unsigned short lB[2][64][40];

    const int tid  = threadIdx.x;
    const int lane = tid & 31;
    const int wave = tid >> 5;             // 0..7
    const int wmo  = (wave >> 1) * 32;     // wave row offset in block tile
    const int wno  = (wave & 1)  * 32;     // wave col offset in block tile
    const int m0   = blockIdx.x * 128;
    const int n0   = blockIdx.y * 64;

    // global->LDS staging assignment:
    //   A tile 128x32: thread t loads 16 bf16: row t>>1, cols (t&1)*16..
    //   B tile  64x32: thread t loads  8 bf16: row t>>2, cols (t&3)*8..
    const int rA = tid >> 1, cA = (tid & 1) * 16;
    const int rB = tid >> 2, cB = (tid & 3) * 8;
    const unsigned short* gA = A + (size_t)(m0 + rA) * K + cA;
    const unsigned short* gB = W + (size_t)(n0 + rB) * K + cB;

    // preload tile 0 into buffer 0
    uint4 sa0 = *(const uint4*)(gA);
    uint4 sa1 = *(const uint4*)(gA + 8);
    uint4 sb0 = *(const uint4*)(gB);
    *(uint4*)&lA[0][rA][cA]     = sa0;
    *(uint4*)&lA[0][rA][cA + 8] = sa1;
    *(uint4*)&lB[0][rB][cB]     = sb0;
    __syncthreads();

    v8f c00 = {}, c01 = {}, c10 = {}, c11 = {};
    const int ksteps = K >> 5;

    for (int i = 0; i < ksteps; ++i) {
        const int buf = i & 1;
        // issue next tile's global loads early (overlap with WMMA below)
        if (i + 1 < ksteps) {
            const unsigned short* pA = gA + (size_t)(i + 1) * 32;
            const unsigned short* pB = gB + (size_t)(i + 1) * 32;
            sa0 = *(const uint4*)(pA);
            sa1 = *(const uint4*)(pA + 8);
            sb0 = *(const uint4*)(pB);
        }

        // fragments from LDS
        FragU a0, a1, b0, b1;
        {
            const int mr = wmo + (lane & 15);
            const int hi = lane >> 4;
            a0.u[0] = *(const uint4*)&lA[buf][mr][hi * 8];
            a0.u[1] = *(const uint4*)&lA[buf][mr][16 + hi * 8];
            a1.u[0] = *(const uint4*)&lA[buf][mr + 16][hi * 8];
            a1.u[1] = *(const uint4*)&lA[buf][mr + 16][16 + hi * 8];
            const int nr = wno + (lane & 15);
            const int kb = hi * 8;
            b0.u[0] = *(const uint4*)&lB[buf][nr][kb];
            b0.u[1] = *(const uint4*)&lB[buf][nr][kb + 16];
            b1.u[0] = *(const uint4*)&lB[buf][nr + 16][kb];
            b1.u[1] = *(const uint4*)&lB[buf][nr + 16][kb + 16];
        }
        c00 = wmma_bf16(a0.v, b0.v, c00);
        c01 = wmma_bf16(a0.v, b1.v, c01);
        c10 = wmma_bf16(a1.v, b0.v, c10);
        c11 = wmma_bf16(a1.v, b1.v, c11);

        // commit next tile to the other buffer
        if (i + 1 < ksteps) {
            const int nb = buf ^ 1;
            *(uint4*)&lA[nb][rA][cA]     = sa0;
            *(uint4*)&lA[nb][rA][cA + 8] = sa1;
            *(uint4*)&lB[nb][rB][cB]     = sb0;
        }
        __syncthreads();
    }

    // Epilogue
    v8f accs[4] = { c00, c01, c10, c11 };
    #pragma unroll
    for (int t = 0; t < 4; ++t) {
        const int ti = t >> 1, tj = t & 1;
        const int mB = m0 + wmo + ti * 16 + (lane >> 4) * 8;
        const int n  = n0 + wno + tj * 16 + (lane & 15);
        const float bv = bias ? bias[n] : 0.0f;
        #pragma unroll
        for (int r = 0; r < 8; ++r) {
            const int m = mB + r;
            float v = accs[t][r] + bv;
            if (resid) v += resid[(size_t)m * N + n];
            if (relu)  v = fmaxf(v, 0.0f);
            if (out32) out32[(size_t)m * N + n] = v;
            if (out16) {
                if (t16) out16[(size_t)n * M + m] = f2bf(v);
                else     out16[(size_t)m * N + n] = f2bf(v);
            }
        }
    }
}

// ---------------------------------------------------------------------------
// Elementwise / reduction kernels (f32 math)
// ---------------------------------------------------------------------------
__device__ __forceinline__ float blk_red(float v, int ismax) {
    __shared__ float sm[8];
    #pragma unroll
    for (int off = 16; off > 0; off >>= 1) {
        float o = __shfl_xor(v, off, 32);
        v = ismax ? fmaxf(v, o) : (v + o);
    }
    if ((threadIdx.x & 31) == 0) sm[threadIdx.x >> 5] = v;
    __syncthreads();
    if (threadIdx.x == 0) {
        float t = sm[0];
        for (int i = 1; i < 8; ++i) t = ismax ? fmaxf(t, sm[i]) : (t + sm[i]);
        sm[0] = t;
    }
    __syncthreads();
    float r = sm[0];
    __syncthreads();
    return r;
}

__global__ __launch_bounds__(256) void cvt_f32_bf16_kernel(
    const float* __restrict__ in, unsigned short* __restrict__ out, int n)
{
    int i = blockIdx.x * 256 + threadIdx.x;
    if (i < n) out[i] = f2bf(in[i]);
}

// one block per row, row softmax (in place)
__global__ __launch_bounds__(256) void softmax_rows_kernel(float* __restrict__ S, int N)
{
    float* p = S + (size_t)blockIdx.x * N;
    const int tid = threadIdx.x;
    float mx = -3.402823466e38f;
    for (int j = tid; j < N; j += 256) mx = fmaxf(mx, p[j]);
    mx = blk_red(mx, 1);
    float sum = 0.0f;
    for (int j = tid; j < N; j += 256) { float e = __expf(p[j] - mx); p[j] = e; sum += e; }
    sum = blk_red(sum, 0);
    const float inv = 1.0f / sum;
    for (int j = tid; j < N; j += 256) p[j] *= inv;
}

// column sums of S[N,N]
__global__ __launch_bounds__(256) void col_sum_kernel(
    const float* __restrict__ S, float* __restrict__ cs, int N)
{
    const int c = blockIdx.x * 256 + threadIdx.x;
    float s = 0.0f;
    for (int r = 0; r < N; ++r) s += S[(size_t)r * N + c];
    cs[c] = s;
}

// S16[i] = bf16( S[i] / (eps + colsum[col]) ), N power of two
__global__ __launch_bounds__(256) void renorm_bf16_kernel(
    const float* __restrict__ S, const float* __restrict__ cs,
    unsigned short* __restrict__ S16, int N)
{
    size_t i = (size_t)blockIdx.x * 256 + threadIdx.x;
    int c = (int)(i & (size_t)(N - 1));
    S16[i] = f2bf(S[i] / (1e-9f + cs[c]));
}

// per-row LayerNorm of (res + sc), D = 512, 256 threads -> 2 elems each
__global__ __launch_bounds__(256) void add_ln_kernel(
    const float* __restrict__ res, const float* __restrict__ sc,
    const float* __restrict__ g, const float* __restrict__ b,
    float* __restrict__ o32, unsigned short* __restrict__ o16, int D)
{
    const int row = blockIdx.x, tid = threadIdx.x;
    const size_t base = (size_t)row * D;
    const int j0 = tid, j1 = tid + 256;
    float x0 = res[base + j0] + sc[base + j0];
    float x1 = res[base + j1] + sc[base + j1];
    float mu = blk_red(x0 + x1, 0) / (float)D;
    float d0 = x0 - mu, d1 = x1 - mu;
    float var = blk_red(d0 * d0 + d1 * d1, 0) / (float)D;
    float inv = rsqrtf(var + 1e-6f);
    float y0 = d0 * inv * g[j0] + b[j0];
    float y1 = d1 * inv * g[j1] + b[j1];
    o32[base + j0] = y0;  o32[base + j1] = y1;
    o16[base + j0] = f2bf(y0);  o16[base + j1] = f2bf(y1);
}

// ---------------------------------------------------------------------------
// Host orchestration
// ---------------------------------------------------------------------------
static inline char* bump(char*& p, size_t bytes) {
    char* r = p;
    p += (bytes + 255) & ~(size_t)255;
    return r;
}

extern "C" void kernel_launch(void* const* d_in, const int* in_sizes, int n_in,
                              void* d_out, int out_size, void* d_ws, size_t ws_size,
                              hipStream_t stream) {
    (void)in_sizes; (void)n_in; (void)out_size; (void)ws_size;
    const int N = NTOK, D = DIM;
    const size_t ND = (size_t)N * D, NNl = (size_t)N * N;

    const float* feat[2] = { (const float*)d_in[0], (const float*)d_in[1] };
    const float* P[2][16];
    for (int br = 0; br < 2; ++br)
        for (int i = 0; i < 16; ++i)
            P[br][i] = (const float*)d_in[2 + br * 16 + i];
    // param index map: 0 pW1,1 pb1,2 pW2,3 pb2,4 rW1,5 rb1,6 rW2,7 rb2,
    //                  8 rW3,9 rb3,10 Wq,11 Wk,12 Wv,13 ln_g,14 ln_b,15 ffnW
    const int widx[9] = {0, 2, 4, 6, 8, 10, 11, 12, 15};

    // ---- workspace carve-up -------------------------------------------------
    char* wp = (char*)d_ws;
    unsigned short *f16[2], *t0[2], *t1[2], *ft[2], *Qb[2], *Kb[2], *Vt[2], *sx16[2];
    unsigned short *w16[2][9];
    float *res32[2], *sc32[2], *s32[2];
    for (int br = 0; br < 2; ++br) {
        f16[br]  = (unsigned short*)bump(wp, ND * 2);
        t0[br]   = (unsigned short*)bump(wp, ND * 2);
        t1[br]   = (unsigned short*)bump(wp, ND * 2);
        ft[br]   = (unsigned short*)bump(wp, ND * 2);
        Qb[br]   = (unsigned short*)bump(wp, ND * 2);
        Kb[br]   = (unsigned short*)bump(wp, ND * 2);
        Vt[br]   = (unsigned short*)bump(wp, ND * 2);
        sx16[br] = (unsigned short*)bump(wp, ND * 2);
        res32[br] = (float*)bump(wp, ND * 4);
        sc32[br]  = (float*)bump(wp, ND * 4);
        s32[br]   = (float*)bump(wp, ND * 4);
        for (int i = 0; i < 9; ++i)
            w16[br][i] = (unsigned short*)bump(wp, (size_t)D * D * 2);
    }
    float*          S   = (float*)bump(wp, NNl * 4);          // attention scratch (shared)
    unsigned short* S16 = (unsigned short*)bump(wp, NNl * 2);
    float*          cs  = (float*)bump(wp, (size_t)N * 4);

    auto cvt = [&](const float* src, unsigned short* dst, size_t n) {
        cvt_f32_bf16_kernel<<<(int)((n + 255) / 256), 256, 0, stream>>>(src, dst, (int)n);
    };
    auto gemm = [&](const unsigned short* Ain, const unsigned short* Win,
                    const float* bias, const float* resid,
                    float* o32, unsigned short* o16,
                    int M, int Nn, int Kk, int relu, int trans16) {
        dim3 grid(M / 128, Nn / 64);
        gemm_bf16_kernel<<<grid, 256, 0, stream>>>(Ain, Win, bias, resid, o32, o16,
                                                   M, Nn, Kk, relu, trans16);
    };

    // ---- convert inputs & weights to bf16 ----------------------------------
    for (int br = 0; br < 2; ++br) {
        cvt(feat[br], f16[br], ND);
        for (int i = 0; i < 9; ++i) cvt(P[br][widx[i]], w16[br][i], (size_t)D * D);
    }

    // ---- per-branch dense chain --------------------------------------------
    for (int br = 0; br < 2; ++br) {
        // proj: relu(x@pW1^T+pb1) @ pW2^T + pb2
        gemm(f16[br], w16[br][0], P[br][1], nullptr, nullptr, t0[br], N, D, D, 1, 0);
        gemm(t0[br],  w16[br][1], P[br][3], nullptr, nullptr, ft[br], N, D, D, 0, 0);
        // residual MLP: relu -> relu -> linear
        gemm(ft[br],  w16[br][2], P[br][5], nullptr, nullptr, t0[br], N, D, D, 1, 0);
        gemm(t0[br],  w16[br][3], P[br][7], nullptr, nullptr, t1[br], N, D, D, 1, 0);
        gemm(t1[br],  w16[br][4], P[br][9], nullptr, res32[br], nullptr, N, D, D, 0, 0);
        // Q, K, V  (V stored transposed: Vt[D, N] so attn@V reuses A@W^T form)
        gemm(ft[br],  w16[br][5], nullptr, nullptr, nullptr, Qb[br], N, D, D, 0, 0);
        gemm(ft[br],  w16[br][6], nullptr, nullptr, nullptr, Kb[br], N, D, D, 0, 0);
        gemm(ft[br],  w16[br][7], nullptr, nullptr, nullptr, Vt[br], N, D, D, 0, 1);
    }

    // ---- cross attention (branches sequential, sharing S scratch) ----------
    for (int br = 0; br < 2; ++br) {
        const int other = 1 - br;
        // logits = Q_br @ K_other^T   [N,N]
        gemm(Qb[br], Kb[other], nullptr, nullptr, S, nullptr, N, N, D, 0, 0);
        softmax_rows_kernel<<<N, 256, 0, stream>>>(S, N);
        col_sum_kernel<<<N / 256, 256, 0, stream>>>(S, cs, N);
        renorm_bf16_kernel<<<(int)(NNl / 256), 256, 0, stream>>>(S, cs, S16, N);
        // sc = attn @ V_br = S16 @ Vt_br^T   [N,D]
        gemm(S16, Vt[br], nullptr, nullptr, sc32[br], nullptr, N, D, N, 0, 0);
    }

    // ---- add & norm, residual FFN, final output ----------------------------
    float* dout = (float*)d_out;
    for (int br = 0; br < 2; ++br) {
        add_ln_kernel<<<N, 256, 0, stream>>>(res32[br], sc32[br], P[br][13], P[br][14],
                                             s32[br], sx16[br], D);
        gemm(sx16[br], w16[br][8], nullptr, s32[br], dout + (size_t)br * ND, nullptr,
             N, D, D, 0, 0);
    }
}
